// BinarizedConv1d_90099823935569
// MI455X (gfx1250) — 8-bit matrix operand order repaired and checked by reading, compiled, not run on hardware
//
#include <hip/hip_runtime.h>

typedef __attribute__((ext_vector_type(16))) int   v16i;
typedef __attribute__((ext_vector_type(8)))  float v8f;

#define CIN   128
#define COUT  128
#define KW    5
#define LSEQ  8192
#define NB    16
#define LPAD  (LSEQ + 4)
#define KDIM  (CIN * KW)

#define FP8_P1 0x38
#define FP8_M1 0xB8
#define FP8_Z  0x00

#define W8_OFF  0
#define W8_SIZE (COUT * KDIM)
#define XT_OFF  W8_SIZE
#define XT_SIZE ((size_t)NB * LPAD * CIN)
#define ST_OFF  (XT_OFF + XT_SIZE)

__device__ __forceinline__ unsigned char fp8sign(float v) {
  return v > 0.f ? (unsigned char)FP8_P1 : (v < 0.f ? (unsigned char)FP8_M1 : (unsigned char)FP8_Z);
}

__global__ void bin_weights(const float* __restrict__ w, unsigned char* __restrict__ w8,
                            float* __restrict__ stats) {
  int i = blockIdx.x * 256 + threadIdx.x;
  if (blockIdx.x == 0 && threadIdx.x < 256) stats[threadIdx.x] = 0.f;
  if (i >= COUT * KDIM) return;
  int co  = i / KDIM;
  int r   = i - co * KDIM;
  int tap = r >> 7;
  int ci  = r & 127;
  w8[i] = fp8sign(w[(co * CIN + ci) * KW + tap]);
}

__global__ void pad_zero(unsigned char* __restrict__ xt) {
  int i = blockIdx.x * 256 + threadIdx.x;
  if (i >= NB * 4 * 32) return;
  int c4 = i & 31;
  int r  = (i >> 5) & 3;
  int b  = i >> 7;
  int row = (r < 2) ? r : (r + LSEQ);
  ((unsigned int*)xt)[((size_t)b * LPAD + row) * 32 + c4] = 0u;
}

__global__ __launch_bounds__(256) void bin_x(const float* __restrict__ x,
                                             unsigned char* __restrict__ xt) {
  __shared__ unsigned char s[64][132];
  int t  = threadIdx.x;
  int b  = blockIdx.x >> 7;
  int l0 = (blockIdx.x & 127) << 6;

  int l_local = t & 63;
  int ci_hi   = t >> 6;
#pragma unroll
  for (int it = 0; it < 32; ++it) {
    int ci = it * 4 + ci_hi;
    float v = x[((size_t)(b * CIN + ci)) * LSEQ + l0 + l_local];
    s[l_local][ci] = fp8sign(v);
  }
  __syncthreads();

  int c4   = t & 31;
  int lrow = t >> 5;
#pragma unroll
  for (int p = 0; p < 8; ++p) {
    int l = p * 8 + lrow;
    unsigned int pack = *(const unsigned int*)&s[l][c4 * 4];
    ((unsigned int*)xt)[((size_t)b * LPAD + l0 + l + 2) * 32 + c4] = pack;
  }
}

__device__ __forceinline__ void conv_acc(const unsigned char* __restrict__ w8,
                                         const unsigned char* __restrict__ xt,
                                         int b, int l0, int wave, int half, int l15,
                                         v8f acc[8]) {
  int kA[16], kB[16];
#pragma unroll
  for (int v = 0; v < 16; ++v) {
    int v8 = v & 7;
    kA[v] = ((v8 >> 1) * 16) + ((v8 & 1) * 4) + half * 8 + ((v >> 3) * 64);
    kB[v] = ((v8 >> 1) * 16) + ((v8 & 1) * 4) + half * 8 + ((v >> 3) * 64);
  }

  const unsigned char* arow = w8 + (size_t)(wave * 16 + l15) * KDIM;

#pragma unroll
  for (int nt = 0; nt < 8; ++nt) acc[nt] = (v8f){};

#pragma unroll
  for (int tap = 0; tap < KW; ++tap) {
    v16i a;
#pragma unroll
    for (int v = 0; v < 16; ++v)
      a[v] = *(const int*)(arow + tap * 128 + kA[v]);
#pragma unroll
    for (int nt = 0; nt < 8; ++nt) {
      const unsigned char* brow =
          xt + ((size_t)(b * LPAD + l0 + nt * 16 + l15 + tap) << 7);
      if (tap < KW - 1)
        __builtin_prefetch(brow + 128, 0, 1);
      v16i bm;
#pragma unroll
      for (int v = 0; v < 16; ++v)
        bm[v] = *(const int*)(brow + kB[v]);
      acc[nt] = __builtin_amdgcn_wmma_f32_16x16x128_fp8_fp8(
          a, bm, (short)0, acc[nt], false, false);
    }
  }
}

__global__ __launch_bounds__(256) void conv_stats(
    const unsigned char* __restrict__ w8, const unsigned char* __restrict__ xt,
    const float* __restrict__ bias,
    float* __restrict__ g_sum, float* __restrict__ g_sq) {
  __shared__ float s_sum[COUT];
  __shared__ float s_sq[COUT];
  int tid = threadIdx.x;
  if (tid < COUT) { s_sum[tid] = 0.f; s_sq[tid] = 0.f; }
  __syncthreads();

  int lane = tid & 31, wave = tid >> 5;
  int half = lane >> 4, l15 = lane & 15;
  int b  = blockIdx.x >> 6;
  int l0 = (blockIdx.x & 63) << 7;

  v8f acc[8];
  conv_acc(w8, xt, b, l0, wave, half, l15, acc);

  float bv[8], psum[8], psq[8];
#pragma unroll
  for (int r = 0; r < 8; ++r) {
    bv[r] = bias[wave * 16 + half * 8 + r];
    psum[r] = 0.f; psq[r] = 0.f;
  }
#pragma unroll
  for (int nt = 0; nt < 8; ++nt) {
#pragma unroll
    for (int r = 0; r < 8; ++r) {
      float val = acc[nt][r] + bv[r];
      psum[r] += val;
      psq[r]  += val * val;
    }
  }
#pragma unroll
  for (int r = 0; r < 8; ++r) {
    int co = wave * 16 + half * 8 + r;
    atomicAdd(&s_sum[co], psum[r]);
    atomicAdd(&s_sq[co],  psq[r]);
  }
  __syncthreads();
  if (tid < COUT) {
    atomicAdd(&g_sum[tid], s_sum[tid]);
    atomicAdd(&g_sq[tid],  s_sq[tid]);
  }
}

__global__ void bn_stats(const float* __restrict__ g_sum, const float* __restrict__ g_sq,
                         float* __restrict__ mean, float* __restrict__ scale) {
  int c = threadIdx.x;
  if (c >= COUT) return;
  const float n = (float)((long)NB * LSEQ);
  float m = g_sum[c] / n;
  float v = g_sq[c] / n - m * m;
  mean[c]  = m;
  scale[c] = rsqrtf(v + 1e-5f);
}

__global__ __launch_bounds__(256) void conv_norm(
    const unsigned char* __restrict__ w8, const unsigned char* __restrict__ xt,
    const float* __restrict__ bias, const float* __restrict__ mean,
    const float* __restrict__ scale, float* __restrict__ out) {
  int tid = threadIdx.x;
  int lane = tid & 31, wave = tid >> 5;
  int half = lane >> 4, l15 = lane & 15;
  int b  = blockIdx.x >> 6;
  int l0 = (blockIdx.x & 63) << 7;

  v8f acc[8];
  conv_acc(w8, xt, b, l0, wave, half, l15, acc);

  float bv[8], m[8], sc[8];
#pragma unroll
  for (int r = 0; r < 8; ++r) {
    int co = wave * 16 + half * 8 + r;
    bv[r] = bias[co];
    m[r]  = mean[co];
    sc[r] = scale[co];
  }
#pragma unroll
  for (int nt = 0; nt < 8; ++nt) {
    int l = l0 + nt * 16 + l15;
#pragma unroll
    for (int r = 0; r < 8; ++r) {
      int co = wave * 16 + half * 8 + r;
      out[((size_t)(b * COUT + co)) * LSEQ + l] = (acc[nt][r] + bv[r] - m[r]) * sc[r];
    }
  }
}

extern "C" void kernel_launch(void* const* d_in, const int* in_sizes, int n_in,
                              void* d_out, int out_size, void* d_ws, size_t ws_size,
                              hipStream_t stream) {
  const float* x    = (const float*)d_in[0];
  const float* w    = (const float*)d_in[1];
  const float* bias = (const float*)d_in[2];
  float* out = (float*)d_out;

  unsigned char* ws = (unsigned char*)d_ws;
  unsigned char* w8 = ws + W8_OFF;
  unsigned char* xt = ws + XT_OFF;
  float* stats = (float*)(ws + ST_OFF);
  float* g_sum = stats;
  float* g_sq  = stats + 128;
  float* mean  = stats + 256;
  float* scale = stats + 384;

  bin_weights<<<(COUT * KDIM + 255) / 256, 256, 0, stream>>>(w, w8, stats);
  pad_zero<<<(NB * 4 * 32 + 255) / 256, 256, 0, stream>>>(xt);
  bin_x<<<NB * (LSEQ / 64), 256, 0, stream>>>(x, xt);

  conv_stats<<<NB * (LSEQ / 128), 256, 0, stream>>>(w8, xt, bias, g_sum, g_sq);
  bn_stats<<<1, 128, 0, stream>>>(g_sum, g_sq, mean, scale);
  conv_norm<<<NB * (LSEQ / 128), 256, 0, stream>>>(w8, xt, bias, mean, scale, out);
}
